// ParallelTransformerBlock_70832600645662
// MI455X (gfx1250) — matrix operand order repaired and checked by reading, not compiled, not run on hardware
//
#include <hip/hip_runtime.h>
#include <hip/hip_bf16.h>
#include <math.h>

// ---------------- problem constants ----------------
constexpr int DIMC  = 2048;
constexpr int HEADS = 16;
constexpr int DHEAD = 128;
constexpr int FFI   = 8192;                    // FF_INNER
constexpr int FUSED = HEADS*DHEAD + DHEAD + DHEAD + 2*FFI;  // 18688
constexpr int BATCH = 2;
constexpr int SEQ   = 2048;
constexpr int ROWS  = BATCH * SEQ;             // 4096

// ---------------- types ----------------
typedef __attribute__((ext_vector_type(16))) __bf16 v16bf;
typedef __attribute__((ext_vector_type(8)))  float  v8f;
typedef __attribute__((ext_vector_type(4)))  int    v4i;

union Frag16 { v16bf bf; v4i i[2]; };

#define AS1 __attribute__((address_space(1)))
#define AS3 __attribute__((address_space(3)))

// CDNA5 async global->LDS staging (ASYNCcnt path), guarded so compile can't break
#if defined(__gfx1250__) && __has_builtin(__builtin_amdgcn_global_load_async_to_lds_b128)
#define HAVE_ASYNC_LDS 1
#else
#define HAVE_ASYNC_LDS 0
#endif

#if defined(__gfx1250__) && __has_builtin(__builtin_amdgcn_s_wait_asynccnt)
#define ASYNC_WAIT() __builtin_amdgcn_s_wait_asynccnt(0)
#else
#define ASYNC_WAIT() asm volatile("s_wait_asynccnt 0x0" ::: "memory")
#endif

#if HAVE_ASYNC_LDS
#define STAGE_WAIT() ASYNC_WAIT()
#else
#define STAGE_WAIT()
#endif

__device__ __forceinline__ unsigned short f32_to_bf16(float f) {
  union { float f; unsigned u; } v; v.f = f;
  unsigned u = v.u;
  unsigned r = u + 0x7FFFu + ((u >> 16) & 1u);   // round-to-nearest-even
  return (unsigned short)(r >> 16);
}
__device__ __forceinline__ float bf16_to_f32(unsigned short h) {
  union { unsigned u; float f; } v; v.u = (unsigned)h << 16; return v.f;
}

// ---------------- LayerNorm -> bf16 ----------------
__global__ __launch_bounds__(256)
void ln_kernel(const float* __restrict__ x, const float* __restrict__ gamma,
               unsigned short* __restrict__ xn) {
  const int row = blockIdx.x;
  const float* xr = x + (size_t)row * DIMC;
  float vals[8], s = 0.f, ss = 0.f;
#pragma unroll
  for (int c = 0; c < 8; ++c) {
    float v = xr[c*256 + threadIdx.x];
    vals[c] = v; s += v; ss += v*v;
  }
#pragma unroll
  for (int off = 1; off < 32; off <<= 1) {
    s  += __shfl_xor(s,  off, 32);
    ss += __shfl_xor(ss, off, 32);
  }
  __shared__ float rs[8], rss[8];
  const int wid = threadIdx.x >> 5, lane = threadIdx.x & 31;
  if (lane == 0) { rs[wid] = s; rss[wid] = ss; }
  __syncthreads();
  s = 0.f; ss = 0.f;
#pragma unroll
  for (int w = 0; w < 8; ++w) { s += rs[w]; ss += rss[w]; }
  const float mean = s * (1.f/2048.f);
  const float var  = ss * (1.f/2048.f) - mean*mean;
  const float rstd = rsqrtf(var + 1e-5f);
#pragma unroll
  for (int c = 0; c < 8; ++c) {
    int col = c*256 + threadIdx.x;
    xn[(size_t)row*DIMC + col] = f32_to_bf16((vals[c]-mean)*rstd*gamma[col]);
  }
}

// ---------------- f32 -> bf16 convert ----------------
__global__ __launch_bounds__(256)
void cvt_kernel(const float* __restrict__ src, unsigned short* __restrict__ dst, size_t n) {
  size_t i = (size_t)blockIdx.x * 256 + threadIdx.x;
  if (i < n) dst[i] = f32_to_bf16(src[i]);
}

// ---------------- bf16 WMMA GEMM ----------------
// MODE: 0 = store f32, 1 = accumulate into f32, 2 = store bf16
// block tile 128x128x64, 8 waves, each wave 32x64 (2x4 fragments of 16x16)
template<int MODE>
__global__ __launch_bounds__(256)
void gemm_bf16_kernel(const unsigned short* __restrict__ A,
                      const unsigned short* __restrict__ Bw,
                      void* __restrict__ Cv,
                      int M, int Nn, int K) {
  constexpr int BK = 64;
  __shared__ __align__(16) unsigned short lA[128][BK + 8];   // [m][k]
  __shared__ __align__(16) unsigned short lB[128][BK + 8];   // [n][k] (transposed)

  const int tid  = threadIdx.x;
  const int wid  = tid >> 5;
  const int lane = tid & 31;
  const int ln15 = lane & 15;
  const int hi   = lane >> 4;          // 0/1
  const int wm   = wid & 3;            // wave row (4 in M)
  const int wn   = wid >> 2;           // wave col (2 in N)
  const int m0   = blockIdx.y * 128;
  const int n0   = blockIdx.x * 128;

  v8f acc[2][4] = {};

  for (int k0 = 0; k0 < K; k0 += BK) {
    // ---- stage A tile: 128x64 halves = 1024 16B chunks, 4 per thread ----
#pragma unroll
    for (int c = 0; c < 4; ++c) {
      int lin = tid + c*256;
      int row = lin >> 3;
      int kc  = (lin & 7) * 8;
      const unsigned short* gp = A + (size_t)(m0+row)*K + k0 + kc;
#if HAVE_ASYNC_LDS
      __builtin_amdgcn_global_load_async_to_lds_b128(
          (AS1 v4i*)(v4i*)gp, (AS3 v4i*)(v4i*)&lA[row][kc], 0, 0);
#else
      *reinterpret_cast<v4i*>(&lA[row][kc]) = *reinterpret_cast<const v4i*>(gp);
#endif
    }
    // ---- stage B tile transposed: [k][n] global -> [n][k] LDS ----
#pragma unroll
    for (int c = 0; c < 4; ++c) {
      int lin = tid + c*256;
      int kr  = lin >> 4;
      int nc  = (lin & 15) * 8;
      v4i g = *reinterpret_cast<const v4i*>(Bw + (size_t)(k0+kr)*Nn + n0 + nc);
      const unsigned short* h = reinterpret_cast<const unsigned short*>(&g);
#pragma unroll
      for (int e = 0; e < 8; ++e) lB[nc+e][kr] = h[e];
    }
    // ---- prefetch next K-stage cache lines (overlaps with WMMA compute) ----
    if (k0 + BK < K) {
      if (tid < 128) {
        __builtin_prefetch(A + (size_t)(m0+tid)*K + (k0 + BK), 0, 1);
      } else {
        int t  = tid - 128;
        int kr = t >> 1;
        int nc = (t & 1) * 64;
        __builtin_prefetch(Bw + (size_t)(k0+BK+kr)*Nn + n0 + nc, 0, 1);
      }
    }
    STAGE_WAIT();
    __syncthreads();

#pragma unroll
    for (int ks = 0; ks < BK; ks += 32) {
      v16bf af[2];
#pragma unroll
      for (int i = 0; i < 2; ++i) {
        int row = wm*32 + i*16 + ln15;
        int kb  = ks + (hi ? 8 : 0);         // A layout: K0-7/K16-23 vs K8-15/K24-31
        Frag16 f;
        f.i[0] = *reinterpret_cast<const v4i*>(&lA[row][kb]);
        f.i[1] = *reinterpret_cast<const v4i*>(&lA[row][kb + 16]);
        af[i] = f.bf;
      }
#pragma unroll
      for (int j = 0; j < 4; ++j) {
        int col = wn*64 + j*16 + ln15;
        int kb  = ks + (hi ? 8 : 0);         // B layout: K0-7/K16-23 vs K8-15/K24-31, lane = N
        Frag16 f;
        f.i[0] = *reinterpret_cast<const v4i*>(&lB[col][kb]);
        f.i[1] = *reinterpret_cast<const v4i*>(&lB[col][kb + 16]);
        v16bf bf = f.bf;
#pragma unroll
        for (int i = 0; i < 2; ++i)
          acc[i][j] = __builtin_amdgcn_wmma_f32_16x16x32_bf16(
              false, af[i], false, bf, (short)0, acc[i][j], false, false);
      }
    }
    __syncthreads();
  }

  // epilogue: C layout VGPR vg -> row vg + 8*hi, lane -> col
#pragma unroll
  for (int i = 0; i < 2; ++i)
#pragma unroll
    for (int j = 0; j < 4; ++j) {
      int col = n0 + wn*64 + j*16 + ln15;
#pragma unroll
      for (int vg = 0; vg < 8; ++vg) {
        int row = m0 + wm*32 + i*16 + vg + 8*hi;
        size_t idx = (size_t)row * Nn + col;
        float v = acc[i][j][vg];
        if (MODE == 2) {
          ((unsigned short*)Cv)[idx] = f32_to_bf16(v);
        } else if (MODE == 1) {
          ((float*)Cv)[idx] += v;
        } else {
          ((float*)Cv)[idx] = v;
        }
      }
    }
}

// ---------------- RoPE on q (xpos scale & d^-0.5), [b,h,n,d] bf16 ----------------
__global__ __launch_bounds__(256)
void rope_q_kernel(const unsigned short* __restrict__ proj, unsigned short* __restrict__ Qo) {
  size_t idx = (size_t)blockIdx.x * 256 + threadIdx.x;   // ROWS*HEADS*64 threads
  int i   = (int)(idx & 63);
  int h   = (int)((idx >> 6) & 15);
  int row = (int)(idx >> 10);
  int b   = row >> 11;
  int nn  = row & (SEQ - 1);
  const unsigned short* qp = proj + (size_t)row * FUSED + h * DHEAD;
  const float qs = 0.08838834764831845f;                 // 128^-0.5
  float q1 = bf16_to_f32(qp[i])      * qs;
  float q2 = bf16_to_f32(qp[i + 64]) * qs;
  float t    = (float)nn;
  float invf = __powf(10000.f, -(float)(2*i) * (1.f/128.f));
  float ang  = t * invf;
  float cs = __cosf(ang), sn = __sinf(ang);
  float bs = ((float)(2*i) + 51.2f) * (1.f/179.2f);
  float pw = (t - 1024.f) * (1.f/512.f);
  float sc = __powf(bs, pw);
  unsigned short* ob = Qo + (((size_t)(b*HEADS + h))*SEQ + nn) * DHEAD;
  ob[i]      = f32_to_bf16((q1*cs - q2*sn) * sc);
  ob[i + 64] = f32_to_bf16((q2*cs + q1*sn) * sc);
}

// ---------------- RoPE on k (1/scale) + v convert, [b,n,d] bf16 ----------------
__global__ __launch_bounds__(256)
void rope_kv_kernel(const unsigned short* __restrict__ proj,
                    unsigned short* __restrict__ Ko, unsigned short* __restrict__ Vo) {
  size_t idx = (size_t)blockIdx.x * 256 + threadIdx.x;   // ROWS*64 threads
  int i   = (int)(idx & 63);
  int row = (int)(idx >> 6);
  int nn  = row & (SEQ - 1);
  const unsigned short* kp = proj + (size_t)row * FUSED + HEADS*DHEAD;
  const unsigned short* vp = kp + DHEAD;
  float k1 = bf16_to_f32(kp[i]), k2 = bf16_to_f32(kp[i + 64]);
  float t    = (float)nn;
  float invf = __powf(10000.f, -(float)(2*i) * (1.f/128.f));
  float ang  = t * invf;
  float cs = __cosf(ang), sn = __sinf(ang);
  float bs = ((float)(2*i) + 51.2f) * (1.f/179.2f);
  float pw = (t - 1024.f) * (1.f/512.f);
  float sc = 1.f / __powf(bs, pw);                       // inverse xpos scale
  unsigned short* kb = Ko + (size_t)row * DHEAD;
  kb[i]      = f32_to_bf16((k1*cs - k2*sn) * sc);
  kb[i + 64] = f32_to_bf16((k2*cs + k1*sn) * sc);
  unsigned short* vb = Vo + (size_t)row * DHEAD;
  vb[i]      = vp[i];
  vb[i + 64] = vp[i + 64];
}

// ---------------- SwiGLU: silu(gate)*x -> bf16 [ROWS, FFI] ----------------
__global__ __launch_bounds__(256)
void ffact_kernel(const unsigned short* __restrict__ proj, unsigned short* __restrict__ F) {
  size_t idx = (size_t)blockIdx.x * 256 + threadIdx.x;   // ROWS*FFI threads
  size_t row = idx >> 13;
  int j = (int)(idx & (FFI - 1));
  const unsigned short* base = proj + row * FUSED + (HEADS*DHEAD + 2*DHEAD);
  float xv = bf16_to_f32(base[j]);
  float g  = bf16_to_f32(base[j + FFI]);
  float act = g / (1.f + __expf(-g)) * xv;
  F[row * FFI + j] = f32_to_bf16(act);
}

// ---------------- flash attention (multi-query, causal) ----------------
// grid (SEQ/128, HEADS, BATCH); 8 waves; wave w: q rows qt*128 + w*16
__global__ __launch_bounds__(256)
void attn_kernel(const unsigned short* __restrict__ Q,   // [B][H][SEQ][128]
                 const unsigned short* __restrict__ Kk,  // [B][SEQ][128]
                 const unsigned short* __restrict__ V,   // [B][SEQ][128]
                 unsigned short* __restrict__ O) {       // [B][SEQ][H*128]
  __shared__ __align__(16) unsigned short sK [64][136];  // [key][d]
  __shared__ __align__(16) unsigned short sVt[128][72];  // [d][key]
  __shared__ __align__(16) unsigned short sP [8][16][72];// per wave [m][k]

  const int tid  = threadIdx.x;
  const int wid  = tid >> 5;
  const int lane = tid & 31;
  const int ln15 = lane & 15;
  const int hi   = lane >> 4;
  const int qt   = blockIdx.x;
  const int h    = blockIdx.y;
  const int b    = blockIdx.z;
  const int qrow0 = qt*128 + wid*16;

  // load q fragments (A layout, K = d = 128 -> 4 chunks)
  v16bf qf[4];
  {
    const unsigned short* qbase =
        Q + (((size_t)(b*HEADS + h))*SEQ + qrow0 + ln15) * DHEAD;
#pragma unroll
    for (int c = 0; c < 4; ++c) {
      int kb = 32*c + (hi ? 8 : 0);
      Frag16 f;
      f.i[0] = *reinterpret_cast<const v4i*>(qbase + kb);
      f.i[1] = *reinterpret_cast<const v4i*>(qbase + kb + 16);
      qf[c] = f.bf;
    }
  }

  v8f o[8] = {};
  float mprev[8], lsum[8];
#pragma unroll
  for (int r = 0; r < 8; ++r) { mprev[r] = -1e30f; lsum[r] = 0.f; }

  const int ntiles = qt*2 + 2;              // keys up to qt*128+127
  for (int jt = 0; jt < ntiles; ++jt) {
    const int j0 = jt * 64;
    // stage K tile [64][128] straight (async direct-to-LDS on CDNA5)
#pragma unroll
    for (int c = 0; c < 4; ++c) {
      int lin = tid + c*256;
      int key = lin >> 4;
      int dc  = (lin & 15) * 8;
      const unsigned short* gp = Kk + ((size_t)b*SEQ + j0 + key)*DHEAD + dc;
#if HAVE_ASYNC_LDS
      __builtin_amdgcn_global_load_async_to_lds_b128(
          (AS1 v4i*)(v4i*)gp, (AS3 v4i*)(v4i*)&sK[key][dc], 0, 0);
#else
      *reinterpret_cast<v4i*>(&sK[key][dc]) = *reinterpret_cast<const v4i*>(gp);
#endif
    }
    // stage V tile transposed [d][key]
#pragma unroll
    for (int c = 0; c < 4; ++c) {
      int lin = tid + c*256;
      int key = lin >> 4;
      int dc  = (lin & 15) * 8;
      v4i g = *reinterpret_cast<const v4i*>(V + ((size_t)b*SEQ + j0 + key)*DHEAD + dc);
      const unsigned short* hh = reinterpret_cast<const unsigned short*>(&g);
#pragma unroll
      for (int e = 0; e < 8; ++e) sVt[dc+e][key] = hh[e];
    }
    STAGE_WAIT();
    __syncthreads();

    // S = q . k^T : 4 fragments of 16 keys, K-dim = d = 128
    v8f s[4] = {};
#pragma unroll
    for (int jj = 0; jj < 4; ++jj) {
      int kl = jj*16 + ln15;                  // key index (B lane = N = key)
#pragma unroll
      for (int c = 0; c < 4; ++c) {
        int db = 32*c + (hi ? 8 : 0);
        Frag16 f;
        f.i[0] = *reinterpret_cast<const v4i*>(&sK[kl][db]);
        f.i[1] = *reinterpret_cast<const v4i*>(&sK[kl][db + 16]);
        s[jj] = __builtin_amdgcn_wmma_f32_16x16x32_bf16(
            false, qf[c], false, f.bf, (short)0, s[jj], false, false);
      }
    }

    // causal mask + online softmax (row = vg + 8*hi within wave q-tile)
#pragma unroll
    for (int vg = 0; vg < 8; ++vg) {
      int row = qrow0 + vg + 8*hi;
      float mt = -1e30f;
#pragma unroll
      for (int jj = 0; jj < 4; ++jj) {
        int col = j0 + jj*16 + ln15;
        float sv = s[jj][vg];
        if (col > row) sv = -1e30f;
        s[jj][vg] = sv;
        mt = fmaxf(mt, sv);
      }
#pragma unroll
      for (int off = 1; off < 16; off <<= 1)
        mt = fmaxf(mt, __shfl_xor(mt, off, 32));
      float mn = fmaxf(mprev[vg], mt);
      float sc = __expf(mprev[vg] - mn);
      lsum[vg] *= sc;
#pragma unroll
      for (int nf = 0; nf < 8; ++nf) o[nf][vg] *= sc;
      float rs = 0.f;
#pragma unroll
      for (int jj = 0; jj < 4; ++jj) {
        float p = __expf(s[jj][vg] - mn);
        s[jj][vg] = p;
        rs += p;
      }
#pragma unroll
      for (int off = 1; off < 16; off <<= 1)
        rs += __shfl_xor(rs, off, 32);
      lsum[vg] += rs;
      mprev[vg] = mn;
      // P (C-layout) -> LDS [m][k] for A-layout reload
#pragma unroll
      for (int jj = 0; jj < 4; ++jj)
        sP[wid][vg + 8*hi][jj*16 + ln15] = f32_to_bf16(s[jj][vg]);
    }

    // O += P . V  (K = 64 keys -> 2 chunks; N = d -> 8 fragments)
#pragma unroll
    for (int kc = 0; kc < 2; ++kc) {
      int kb0 = 32*kc + (hi ? 8 : 0);
      Frag16 pa;
      pa.i[0] = *reinterpret_cast<const v4i*>(&sP[wid][ln15][kb0]);
      pa.i[1] = *reinterpret_cast<const v4i*>(&sP[wid][ln15][kb0 + 16]);
      int kb = 32*kc + (hi ? 8 : 0);
#pragma unroll
      for (int nf = 0; nf < 8; ++nf) {
        int d = nf*16 + ln15;
        Frag16 vb;
        vb.i[0] = *reinterpret_cast<const v4i*>(&sVt[d][kb]);
        vb.i[1] = *reinterpret_cast<const v4i*>(&sVt[d][kb + 16]);
        o[nf] = __builtin_amdgcn_wmma_f32_16x16x32_bf16(
            false, pa.bf, false, vb.bf, (short)0, o[nf], false, false);
      }
    }
    __syncthreads();
  }

  // normalize + store to [b][n][h*128] bf16
#pragma unroll
  for (int vg = 0; vg < 8; ++vg) {
    float inv = 1.f / lsum[vg];
    int row = qrow0 + vg + 8*hi;
    unsigned short* ob = O + ((size_t)b*SEQ + row) * (HEADS*DHEAD) + h*DHEAD;
#pragma unroll
    for (int nf = 0; nf < 8; ++nf)
      ob[nf*16 + ln15] = f32_to_bf16(o[nf][vg] * inv);
  }
}

// ---------------- launcher ----------------
extern "C" void kernel_launch(void* const* d_in, const int* in_sizes, int n_in,
                              void* d_out, int out_size, void* d_ws, size_t ws_size,
                              hipStream_t stream) {
  (void)in_sizes; (void)n_in; (void)out_size; (void)ws_size;
  const float* x       = (const float*)d_in[0];
  const float* gamma   = (const float*)d_in[1];
  const float* w_fused = (const float*)d_in[2];
  const float* w_attn  = (const float*)d_in[3];
  const float* w_ff    = (const float*)d_in[4];
  float* out = (float*)d_out;

  char* ws = (char*)d_ws;
  size_t o = 0;
  unsigned short* xn   = (unsigned short*)(ws + o); o += (size_t)ROWS * DIMC * 2;
  unsigned short* wf   = (unsigned short*)(ws + o); o += (size_t)DIMC * FUSED * 2;
  unsigned short* wao  = (unsigned short*)(ws + o); o += (size_t)DIMC * DIMC * 2;
  unsigned short* wff  = (unsigned short*)(ws + o); o += (size_t)FFI  * DIMC * 2;
  unsigned short* proj = (unsigned short*)(ws + o); o += (size_t)ROWS * FUSED * 2;
  unsigned short* q    = (unsigned short*)(ws + o); o += (size_t)ROWS * DIMC * 2;
  unsigned short* kk   = (unsigned short*)(ws + o); o += (size_t)ROWS * DHEAD * 2;
  unsigned short* vv   = (unsigned short*)(ws + o); o += (size_t)ROWS * DHEAD * 2;
  unsigned short* fact = (unsigned short*)(ws + o); o += (size_t)ROWS * FFI * 2;
  unsigned short* ao   = (unsigned short*)(ws + o); o += (size_t)ROWS * DIMC * 2;

  // 1) layernorm + weight downconversion
  ln_kernel<<<ROWS, 256, 0, stream>>>(x, gamma, xn);
  {
    size_t n1 = (size_t)DIMC * FUSED;
    cvt_kernel<<<(unsigned)((n1 + 255) / 256), 256, 0, stream>>>(w_fused, wf, n1);
    size_t n2 = (size_t)DIMC * DIMC;
    cvt_kernel<<<(unsigned)((n2 + 255) / 256), 256, 0, stream>>>(w_attn, wao, n2);
    size_t n3 = (size_t)FFI * DIMC;
    cvt_kernel<<<(unsigned)((n3 + 255) / 256), 256, 0, stream>>>(w_ff, wff, n3);
  }

  // 2) fused projection GEMM: [4096,2048] x [2048,18688] -> bf16 proj
  gemm_bf16_kernel<2><<<dim3(FUSED/128, ROWS/128), 256, 0, stream>>>(
      xn, wf, (void*)proj, ROWS, FUSED, DIMC);

  // 3) rope / swiglu postprocessing
  rope_q_kernel<<<(ROWS*HEADS*64)/256, 256, 0, stream>>>(proj, q);
  rope_kv_kernel<<<(ROWS*64)/256, 256, 0, stream>>>(proj, kk, vv);
  ffact_kernel<<<(unsigned)(((size_t)ROWS*FFI)/256), 256, 0, stream>>>(proj, fact);

  // 4) causal multi-query flash attention
  attn_kernel<<<dim3(SEQ/128, HEADS, BATCH), 256, 0, stream>>>(q, kk, vv, ao);

  // 5) output GEMMs: attn-out writes, FF-out accumulates
  gemm_bf16_kernel<0><<<dim3(DIMC/128, ROWS/128), 256, 0, stream>>>(
      ao, wao, (void*)out, ROWS, DIMC, DIMC);
  gemm_bf16_kernel<1><<<dim3(DIMC/128, ROWS/128), 256, 0, stream>>>(
      fact, wff, (void*)out, ROWS, DIMC, FFI);
}
